// GPT_17248588661276
// MI455X (gfx1250) — compile-verified, matrix operand order checked by reading, not run on hardware
//
#include <hip/hip_runtime.h>
#include <hip/hip_bf16.h>

typedef __bf16 bf16;
typedef __attribute__((ext_vector_type(16))) __bf16 v16bf;
typedef __attribute__((ext_vector_type(8)))  float  v8f;

#define DEV __device__ __forceinline__

// ---------- scalar helpers ----------
DEV bf16 f2bf(float f) {
    unsigned u = __builtin_bit_cast(unsigned, f);
    unsigned r = (u + 0x7FFFu + ((u >> 16) & 1u)) >> 16;   // round-to-nearest-even
    unsigned short s = (unsigned short)r;
    return __builtin_bit_cast(bf16, s);
}
DEV float gelu_tanh(float x) {
    const float c = 0.7978845608028654f;
    float t = c * (x + 0.044715f * x * x * x);
    return 0.5f * x * (1.0f + tanhf(t));
}
DEV v8f wmma_bf16(v16bf a, v16bf b, v8f c) {
    // D = A(16x32 bf16) * B(32x16 bf16) + C(16x16 f32)
    return __builtin_amdgcn_wmma_f32_16x16x32_bf16(false, a, false, b, (short)0, c, false, false);
}

// ---------- model constants ----------
#define BB   4
#define TT   1024
#define MM   4096        // B*T
#define CC   768
#define HH   2304        // 3C
#define FF   3072        // 4C
#define VV   50257
#define NHEAD 12
#define HD   64
#define LAYERS 12
#define BLK_SIZE 512

// =====================================================================
// elementwise convert f32 -> bf16
__global__ void cvt_f32_bf16(const float* __restrict__ s, bf16* __restrict__ d, size_t n) {
    size_t i = (size_t)blockIdx.x * blockDim.x + threadIdx.x;
    if (i < n) d[i] = f2bf(s[i]);
}

// transpose-convert: src [K][N] f32 row-major -> dst [N][K] bf16 (so WMMA B-fragments are contiguous)
__global__ void cvtT_f32_bf16(const float* __restrict__ s, bf16* __restrict__ d, int K, int N) {
    size_t i = (size_t)blockIdx.x * blockDim.x + threadIdx.x;
    if (i >= (size_t)K * N) return;
    int k = (int)(i / N), n = (int)(i % N);
    d[(size_t)n * K + k] = f2bf(s[i]);
}

// token embedding gather: x[m][c] = wte[idx[m]][c]
__global__ void embed_k(const int* __restrict__ idx, const float* __restrict__ wte,
                        float* __restrict__ x) {
    size_t i = (size_t)blockIdx.x * blockDim.x + threadIdx.x;
    if (i >= (size_t)MM * CC) return;
    int m = (int)(i / CC), c = (int)(i % CC);
    x[i] = wte[(size_t)idx[m] * CC + c];
}

// LayerNorm: one block per row; write bf16 (feeds WMMA A)
__global__ __launch_bounds__(256) void ln_k(const float* __restrict__ x,
                                            const float* __restrict__ g,
                                            const float* __restrict__ bta,
                                            bf16* __restrict__ h) {
    __shared__ float red[256];
    const int m = blockIdx.x, tid = threadIdx.x;
    const float* r = x + (size_t)m * CC;
    float s = 0.f;
    for (int c = tid; c < CC; c += 256) s += r[c];
    red[tid] = s; __syncthreads();
    for (int o = 128; o > 0; o >>= 1) { if (tid < o) red[tid] += red[tid + o]; __syncthreads(); }
    float mean = red[0] * (1.0f / CC);
    __syncthreads();
    float s2 = 0.f;
    for (int c = tid; c < CC; c += 256) { float d = r[c] - mean; s2 += d * d; }
    red[tid] = s2; __syncthreads();
    for (int o = 128; o > 0; o >>= 1) { if (tid < o) red[tid] += red[tid + o]; __syncthreads(); }
    float rstd = rsqrtf(red[0] * (1.0f / CC) + 1e-5f);
    for (int c = tid; c < CC; c += 256)
        h[(size_t)m * CC + c] = f2bf((r[c] - mean) * rstd * g[c] + bta[c]);
}

// =====================================================================
// WMMA GEMM: C[M][N] = A[M][K](bf16,row-major) * Bt[N][K](bf16, B transposed) + bias
// flags: 1=GELU, 2=add f32 residual (stride N), 4=write f32 (else bf16)
// Each wave owns a 32(M) x 64(N) tile: 2x4 C-tiles.
// K loop unrolled by 64 into two phases with DISJOINT fragment registers
// (written only by loads -> no v_mov rotation): phase1 loads are issued
// before phase0's WMMAs, so waits become s_wait_loadcnt >0 instead of 0.
// Requires K % 64 == 0 (true: 768, 3072). Last phase-0 reload overreads
// <=64B past the logical end; all workspace allocations are padded 256B.
__global__ __launch_bounds__(256) void gemm_wmma(const bf16* __restrict__ A,
                                                 const bf16* __restrict__ Bt,
                                                 const float* __restrict__ bias,
                                                 const float* __restrict__ resid,
                                                 bf16* __restrict__ outb,
                                                 float* __restrict__ outf,
                                                 int M, int N, int K, int flags) {
    const int lane = threadIdx.x & 31;
    const int wid  = blockIdx.x * 8 + (threadIdx.x >> 5);
    const int mTiles = M >> 5;
    const int nTiles = (N + 63) >> 6;
    if (wid >= mTiles * nTiles) return;
    const int m0 = (wid % mTiles) << 5;
    const int n0 = (wid / mTiles) << 6;
    const int mr = lane & 15;           // row (A) / column (B) within tile
    const int ks = (lane >> 4) << 4;    // K-half select

    v8f acc[2][4] = {};
    const bf16* a0p = A + (size_t)(m0 + mr) * K + ks;
    const bf16* a1p = A + (size_t)(m0 + 16 + mr) * K + ks;
    const bf16* bp[4];
#pragma unroll
    for (int j = 0; j < 4; j++) {
        int n = n0 + j * 16 + mr; if (n >= N) n = N - 1;   // clamp (edge duplicates never stored)
        bp[j] = Bt + (size_t)n * K + ks;
    }

    // phase registers: only ever written by loads (no rotation copies)
    v16bf A00, A10, B0j0, B0j1, B0j2, B0j3;   // phase 0
    v16bf A01, A11, B1j0, B1j1, B1j2, B1j3;   // phase 1

    // prologue: phase-0 fragments at k=0
    A00  = *(const v16bf*)(a0p);
    A10  = *(const v16bf*)(a1p);
    B0j0 = *(const v16bf*)(bp[0]);
    B0j1 = *(const v16bf*)(bp[1]);
    B0j2 = *(const v16bf*)(bp[2]);
    B0j3 = *(const v16bf*)(bp[3]);

    for (int k = 0; k < K; k += 64) {
        // issue phase-1 loads (k+32) before consuming phase-0
        A01  = *(const v16bf*)(a0p + k + 32);
        A11  = *(const v16bf*)(a1p + k + 32);
        B1j0 = *(const v16bf*)(bp[0] + k + 32);
        B1j1 = *(const v16bf*)(bp[1] + k + 32);
        B1j2 = *(const v16bf*)(bp[2] + k + 32);
        B1j3 = *(const v16bf*)(bp[3] + k + 32);
        __builtin_prefetch(a0p + k + 128, 0, 3);
        __builtin_prefetch(bp[0] + k + 128, 0, 3);
        __builtin_prefetch(bp[2] + k + 128, 0, 3);

        // compute phase 0
        acc[0][0] = wmma_bf16(A00, B0j0, acc[0][0]);
        acc[1][0] = wmma_bf16(A10, B0j0, acc[1][0]);
        acc[0][1] = wmma_bf16(A00, B0j1, acc[0][1]);
        acc[1][1] = wmma_bf16(A10, B0j1, acc[1][1]);
        acc[0][2] = wmma_bf16(A00, B0j2, acc[0][2]);
        acc[1][2] = wmma_bf16(A10, B0j2, acc[1][2]);
        acc[0][3] = wmma_bf16(A00, B0j3, acc[0][3]);
        acc[1][3] = wmma_bf16(A10, B0j3, acc[1][3]);

        // issue phase-0 loads for k+64 (overread on final iter lands in 256B pad)
        A00  = *(const v16bf*)(a0p + k + 64);
        A10  = *(const v16bf*)(a1p + k + 64);
        B0j0 = *(const v16bf*)(bp[0] + k + 64);
        B0j1 = *(const v16bf*)(bp[1] + k + 64);
        B0j2 = *(const v16bf*)(bp[2] + k + 64);
        B0j3 = *(const v16bf*)(bp[3] + k + 64);

        // compute phase 1
        acc[0][0] = wmma_bf16(A01, B1j0, acc[0][0]);
        acc[1][0] = wmma_bf16(A11, B1j0, acc[1][0]);
        acc[0][1] = wmma_bf16(A01, B1j1, acc[0][1]);
        acc[1][1] = wmma_bf16(A11, B1j1, acc[1][1]);
        acc[0][2] = wmma_bf16(A01, B1j2, acc[0][2]);
        acc[1][2] = wmma_bf16(A11, B1j2, acc[1][2]);
        acc[0][3] = wmma_bf16(A01, B1j3, acc[0][3]);
        acc[1][3] = wmma_bf16(A11, B1j3, acc[1][3]);
    }

    // epilogue: C-tile layout: vgpr r -> (m = r + 8*(lane/16), n = lane%16)
#pragma unroll
    for (int i = 0; i < 2; i++) {
#pragma unroll
        for (int j = 0; j < 4; j++) {
            int n = n0 + j * 16 + (lane & 15);
            if (n >= N) continue;
            float bv = bias ? bias[n] : 0.0f;
            int mb = m0 + i * 16 + ((lane >> 4) << 3);
#pragma unroll
            for (int r = 0; r < 8; r++) {
                float v = acc[i][j][r] + bv;
                if (flags & 1) v = gelu_tanh(v);
                size_t o = (size_t)(mb + r) * N + n;
                if (flags & 2) v += resid[o];
                if (flags & 4) outf[o] = v; else outb[o] = f2bf(v);
            }
        }
    }
}

// =====================================================================
// split fused qkv [M][3C] bf16 into Q,K: [B,H,T,64] and V transposed: [B,H,64,T]
__global__ void split_qkv(const bf16* __restrict__ qkv, bf16* __restrict__ Q,
                          bf16* __restrict__ Kd, bf16* __restrict__ Vt) {
    size_t i = (size_t)blockIdx.x * blockDim.x + threadIdx.x;
    if (i >= (size_t)MM * HH) return;
    int m = (int)(i / HH), c = (int)(i % HH);
    int b = m >> 10, t = m & 1023;
    int which = c / CC, hc = c % CC, h = hc >> 6, d = hc & 63;
    bf16 v = qkv[i];
    size_t bh = (size_t)(b * NHEAD + h);
    if (which == 0)      Q [(bh * TT + t) * HD + d] = v;
    else if (which == 1) Kd[(bh * TT + t) * HD + d] = v;
    else                 Vt[(bh * HD + d) * TT + t] = v;
}

// =====================================================================
// Flash-style attention, one wave per (b,h, 16-query tile). No causal mask
// (reference applies plain softmax). scale = sqrt(hd) = 8 (reference MULTIPLIES).
__global__ __launch_bounds__(256) void attn_wmma(const bf16* __restrict__ Q,
                                                 const bf16* __restrict__ Km,
                                                 const bf16* __restrict__ Vt,
                                                 bf16* __restrict__ Y) {
    __shared__ bf16 ldsP[8][16 * 32];           // per-wave P-tile for C->A layout swap
    const int lane  = threadIdx.x & 31;
    const int wslot = threadIdx.x >> 5;
    const int wid   = blockIdx.x * 8 + wslot;
    if (wid >= BB * NHEAD * (TT / 16)) return;
    const int qt = wid & 63;
    const int bh = wid >> 6;
    const int b = bh / NHEAD, h = bh % NHEAD;

    const bf16* q  = Q  + ((size_t)bh * TT + qt * 16) * HD;
    const bf16* kp = Km + (size_t)bh * TT * HD;
    const bf16* vp = Vt + (size_t)bh * HD * TT;

    const int mr = lane & 15;
    const int ks = (lane >> 4) << 4;

    // Q A-fragments for hd 0..31 and 32..63 (kept resident)
    v16bf qa0 = *(const v16bf*)(q + mr * HD + ks);
    v16bf qa1 = *(const v16bf*)(q + mr * HD + 32 + ks);

    v8f o[4] = {};                 // O tile 16 x 64 (4 N-tiles of hd)
    float rmax[8], rsum[8];
#pragma unroll
    for (int r = 0; r < 8; r++) { rmax[r] = -1e30f; rsum[r] = 0.f; }

    for (int kc = 0; kc < TT; kc += 32) {
        // Issue ALL memory for this chunk up front: K^T B-fragments and
        // V B-fragments (V does not depend on the softmax -> overlaps VALU).
        const bf16* kr0 = kp + (size_t)(kc + mr) * HD;
        const bf16* kr1 = kp + (size_t)(kc + 16 + mr) * HD;
        v16bf kb00 = *(const v16bf*)(kr0 + ks);          // tile0, hd 0..31
        v16bf kb01 = *(const v16bf*)(kr0 + 32 + ks);     // tile0, hd 32..63
        v16bf kb10 = *(const v16bf*)(kr1 + ks);          // tile1, hd 0..31
        v16bf kb11 = *(const v16bf*)(kr1 + 32 + ks);     // tile1, hd 32..63
        v16bf vb[4];
#pragma unroll
        for (int j = 0; j < 4; j++)
            vb[j] = *(const v16bf*)(vp + (size_t)(j * 16 + mr) * TT + kc + ks);

        // S = Q * K^T for 32 keys (two 16-key N-tiles), K-dim = hd = 64
        v8f s[2] = {};
        s[0] = wmma_bf16(qa0, kb00, s[0]);
        s[0] = wmma_bf16(qa1, kb01, s[0]);
        s[1] = wmma_bf16(qa0, kb10, s[1]);
        s[1] = wmma_bf16(qa1, kb11, s[1]);

        // online softmax per row (row m striped across a 16-lane half)
#pragma unroll
        for (int r = 0; r < 8; r++) {
            float s0 = s[0][r] * 8.0f, s1 = s[1][r] * 8.0f;
            float mx = fmaxf(s0, s1);
#pragma unroll
            for (int d = 1; d < 16; d <<= 1) mx = fmaxf(mx, __shfl_xor(mx, d, 16));
            float nm   = fmaxf(rmax[r], mx);
            float corr = __expf(rmax[r] - nm);
            float p0 = __expf(s0 - nm), p1 = __expf(s1 - nm);
            float ls = p0 + p1;
#pragma unroll
            for (int d = 1; d < 16; d <<= 1) ls += __shfl_xor(ls, d, 16);
            rsum[r] = rsum[r] * corr + ls;
            rmax[r] = nm;
#pragma unroll
            for (int j = 0; j < 4; j++) o[j][r] *= corr;
            int lrow = ((lane >> 4) << 3) + r;
            ldsP[wslot][lrow * 32 + (lane & 15)]      = f2bf(p0);
            ldsP[wslot][lrow * 32 + 16 + (lane & 15)] = f2bf(p1);
        }
        asm volatile("s_wait_dscnt 0" ::: "memory");   // LDS RAW fence (same-wave DS is in-order; belt & braces)
        // P as A-fragment (16 queries x 32 keys) from LDS; V fragments already resident
        v16bf pa = *(const v16bf*)&ldsP[wslot][mr * 32 + ks];
#pragma unroll
        for (int j = 0; j < 4; j++) o[j] = wmma_bf16(pa, vb[j], o[j]);
    }
    // write Y in [B,T,C] bf16 (feeds proj GEMM)
#pragma unroll
    for (int j = 0; j < 4; j++) {
#pragma unroll
        for (int r = 0; r < 8; r++) {
            int m = ((lane >> 4) << 3) + r;
            float val = o[j][r] / rsum[r];
            Y[((size_t)b * TT + qt * 16 + m) * CC + h * HD + j * 16 + (lane & 15)] = f2bf(val);
        }
    }
}

// =====================================================================
// loss = mean over selected rows of (logsumexp(row) - row[target])
__global__ void zero1(float* p) { if (threadIdx.x == 0 && blockIdx.x == 0) *p = 0.f; }

__global__ __launch_bounds__(256) void loss_k(const float* __restrict__ logits,
                                              const int* __restrict__ idx,
                                              float* __restrict__ loss) {
    __shared__ float red[256];
    const int blk = blockIdx.x, tid = threadIdx.x;
    const int b = blk / BLK_SIZE, ti = blk % BLK_SIZE;
    const size_t m = (size_t)b * TT + (BLK_SIZE - 1) + ti;   // t' in [511, 1022]
    const float* row = logits + m * (size_t)VV;
    const int tgt = idx[b * TT + BLK_SIZE + ti];
    float mx = -1e30f;
    for (int v = tid; v < VV; v += 256) mx = fmaxf(mx, row[v]);
    red[tid] = mx; __syncthreads();
    for (int o = 128; o > 0; o >>= 1) { if (tid < o) red[tid] = fmaxf(red[tid], red[tid + o]); __syncthreads(); }
    float M = red[0]; __syncthreads();
    float s = 0.f;
    for (int v = tid; v < VV; v += 256) s += __expf(row[v] - M);
    red[tid] = s; __syncthreads();
    for (int o = 128; o > 0; o >>= 1) { if (tid < o) red[tid] += red[tid + o]; __syncthreads(); }
    if (tid == 0) {
        float nll = (M + __logf(red[0])) - row[tgt];
        atomicAdd(loss, nll * (1.0f / (BB * BLK_SIZE)));
    }
}

// =====================================================================
extern "C" void kernel_launch(void* const* d_in, const int* in_sizes, int n_in,
                              void* d_out, int out_size, void* d_ws, size_t ws_size,
                              hipStream_t stream) {
    (void)in_sizes; (void)n_in; (void)out_size; (void)ws_size;
    const int*   idx    = (const int*)  d_in[0];
    const float* wte    = (const float*)d_in[1];
    const float* ln1_g  = (const float*)d_in[2];
    const float* ln1_b  = (const float*)d_in[3];
    const float* w_qkv  = (const float*)d_in[4];
    const float* b_qkv  = (const float*)d_in[5];
    const float* w_proj = (const float*)d_in[6];
    const float* b_proj = (const float*)d_in[7];
    const float* ln2_g  = (const float*)d_in[8];
    const float* ln2_b  = (const float*)d_in[9];
    const float* w_fc1  = (const float*)d_in[10];
    const float* b_fc1  = (const float*)d_in[11];
    const float* w_fc2  = (const float*)d_in[12];
    const float* b_fc2  = (const float*)d_in[13];

    float* logits = (float*)d_out;                         // [M][V]
    float* loss   = logits + (size_t)MM * VV;              // 1 element

    // ---- workspace layout (every allocation padded +256B for the
    // pipelined GEMM's trailing phase-0 overread and prefetch-ahead) ----
    char* ws = (char*)d_ws;
    size_t off = 0;
    auto give = [&](size_t bytes) -> char* {
        char* p = ws + off; off += (bytes + 256 + 255) & ~(size_t)255; return p;
    };
    float* x    = (float*)give((size_t)MM * CC * 4);       // residual stream
    bf16*  hb   = (bf16*) give((size_t)MM * CC * 2);       // LN out / x-bf16 (GEMM A)
    bf16*  qkvb = (bf16*) give((size_t)MM * HH * 2);
    bf16*  Qb   = (bf16*) give((size_t)BB * NHEAD * TT * HD * 2);
    bf16*  Kb   = (bf16*) give((size_t)BB * NHEAD * TT * HD * 2);
    bf16*  Vtb  = (bf16*) give((size_t)BB * NHEAD * HD * TT * 2);
    bf16*  Yb   = (bf16*) give((size_t)MM * CC * 2);
    bf16*  mlpb = (bf16*) give((size_t)MM * FF * 2);
    bf16*  qkvT = (bf16*) give((size_t)HH * CC * 2);       // per-layer transposed bf16 weights
    bf16*  prjT = (bf16*) give((size_t)CC * CC * 2);
    bf16*  fc1T = (bf16*) give((size_t)FF * CC * 2);
    bf16*  fc2T = (bf16*) give((size_t)CC * FF * 2);
    bf16*  wteb = (bf16*) give((size_t)VV * CC * 2);       // logits B (already [N][K])

    auto eblocks = [](size_t n) { return (unsigned)((n + 255) / 256); };
    auto gblocks = [](int M, int N) { int w = (M / 32) * ((N + 63) / 64); return (unsigned)((w + 7) / 8); };

    // wte -> bf16 (logits B matrix), and token embedding
    cvt_f32_bf16<<<eblocks((size_t)VV * CC), 256, 0, stream>>>(wte, wteb, (size_t)VV * CC);
    embed_k<<<eblocks((size_t)MM * CC), 256, 0, stream>>>(idx, wte, x);

    for (int l = 0; l < LAYERS; l++) {
        const float* g1  = ln1_g  + (size_t)l * CC;
        const float* be1 = ln1_b  + (size_t)l * CC;
        const float* wq  = w_qkv  + (size_t)l * CC * HH;
        const float* bq  = b_qkv  + (size_t)l * HH;
        const float* wp  = w_proj + (size_t)l * CC * CC;
        const float* bp  = b_proj + (size_t)l * CC;
        const float* g2  = ln2_g  + (size_t)l * CC;
        const float* be2 = ln2_b  + (size_t)l * CC;
        const float* wf1 = w_fc1  + (size_t)l * CC * FF;
        const float* bf1 = b_fc1  + (size_t)l * FF;
        const float* wf2 = w_fc2  + (size_t)l * FF * CC;
        const float* bf2 = b_fc2  + (size_t)l * CC;

        // transpose-convert this layer's weights to bf16 [N][K] (per-layer set fits in L2)
        cvtT_f32_bf16<<<eblocks((size_t)CC * HH), 256, 0, stream>>>(wq,  qkvT, CC, HH);
        cvtT_f32_bf16<<<eblocks((size_t)CC * CC), 256, 0, stream>>>(wp,  prjT, CC, CC);
        cvtT_f32_bf16<<<eblocks((size_t)CC * FF), 256, 0, stream>>>(wf1, fc1T, CC, FF);
        cvtT_f32_bf16<<<eblocks((size_t)FF * CC), 256, 0, stream>>>(wf2, fc2T, FF, CC);

        // LN1 -> h ; qkv = h @ Wq + bq (bf16 out)
        ln_k<<<MM, 256, 0, stream>>>(x, g1, be1, hb);
        gemm_wmma<<<gblocks(MM, HH), 256, 0, stream>>>(hb, qkvT, bq, nullptr, qkvb, nullptr,
                                                       MM, HH, CC, 0);
        // split + V transpose; flash attention
        split_qkv<<<eblocks((size_t)MM * HH), 256, 0, stream>>>(qkvb, Qb, Kb, Vtb);
        attn_wmma<<<(BB * NHEAD * (TT / 16) + 7) / 8, 256, 0, stream>>>(Qb, Kb, Vtb, Yb);
        // x = x + Y @ Wp + bp  (f32 out with residual)
        gemm_wmma<<<gblocks(MM, CC), 256, 0, stream>>>(Yb, prjT, bp, x, nullptr, x,
                                                       MM, CC, CC, 2 | 4);
        // LN2 -> h ; mlp = gelu(h @ Wf1 + bf1) (bf16 out)
        ln_k<<<MM, 256, 0, stream>>>(x, g2, be2, hb);
        gemm_wmma<<<gblocks(MM, FF), 256, 0, stream>>>(hb, fc1T, bf1, nullptr, mlpb, nullptr,
                                                       MM, FF, CC, 1);
        // x = x + mlp @ Wf2 + bf2
        gemm_wmma<<<gblocks(MM, CC), 256, 0, stream>>>(mlpb, fc2T, bf2, x, nullptr, x,
                                                       MM, CC, FF, 2 | 4);
    }

    // logits = x @ wte^T : convert x to bf16, B = wteb ([V][C] == [N][K])
    cvt_f32_bf16<<<eblocks((size_t)MM * CC), 256, 0, stream>>>(x, hb, (size_t)MM * CC);
    gemm_wmma<<<gblocks(MM, VV), 256, 0, stream>>>(hb, wteb, nullptr, nullptr, nullptr, logits,
                                                   MM, VV, CC, 4);

    // loss
    zero1<<<1, 32, 0, stream>>>(loss);
    loss_k<<<BB * BLK_SIZE, 256, 0, stream>>>(logits, idx, loss);
}
